// ReadUnit_11630771437685
// MI455X (gfx1250) — matrix operand order repaired and checked by reading, not compiled, not run on hardware
//
#include <hip/hip_runtime.h>
#include <hip/hip_bf16.h>
#include <math.h>

// ---------------------------------------------------------------------------
// MI455X (gfx1250) fused ReadUnit.
//  * bf16 WMMA (v_wmma_f32_16x16x32_bf16) for the 79-GFLOP GEMM chain
//    (~16us at dense bf16 matrix rate vs ~125us in fp32 16x16x4).
//  * M_TILE=32 with B-fragment reuse across both M-tiles per wave: weight
//    traffic from L2 drops to ~2.7GB total (vs 5GB at M_TILE=16), the
//    dominant cost term given 23.3TB/s HBM / L2-resident 1.5MB weights.
//  * k tile staged into LDS via gfx1250 async DMA
//    (global_load_async_to_lds_b128, ASYNCcnt) instead of VGPR bounce.
//  * fp32 epilogues / softmax / final reduction for accuracy.
// ---------------------------------------------------------------------------

typedef __attribute__((ext_vector_type(16))) __bf16 bf16x16;
typedef __attribute__((ext_vector_type(8)))  __bf16 bf16x8;
typedef __attribute__((ext_vector_type(8)))  float  f32x8;

#define SHUF16(lo, hi) __builtin_shufflevector((lo), (hi), \
    0,1,2,3,4,5,6,7,8,9,10,11,12,13,14,15)

#define NB   256   // batch
#define ND   512   // feature dim
#define NHW  196   // 14*14
#define NK2  1024  // 2*D
#define MT   32    // rows per tile

// ------------------------- workspace layout (bytes) -------------------------
// [0)        Wm fp32   256*512*4  = 524288
// [524288)   m bf16    256*512*2  = 262144
// [786432)   Wddm bf16 512*512*2  = 524288
// [1310720)  Wd2d bf16 512*1024*2 = 1048576
// [2359296)  Wdd bf16  512*512*2  = 524288   (total 2883584 B)

// ---------------------------------------------------------------------------
__global__ void prep_kernel(const float* __restrict__ m,
                            const float* __restrict__ wddm,
                            const float* __restrict__ wd2d,
                            const float* __restrict__ wdd,
                            __bf16* __restrict__ mb,
                            __bf16* __restrict__ wddmb,
                            __bf16* __restrict__ wd2db,
                            __bf16* __restrict__ wddb) {
  int stride = gridDim.x * blockDim.x;
  int tid = blockIdx.x * blockDim.x + threadIdx.x;
  for (int i = tid; i < NB * ND; i += stride)   mb[i]    = (__bf16)m[i];
  for (int i = tid; i < ND * ND; i += stride)   wddmb[i] = (__bf16)wddm[i];
  for (int i = tid; i < ND * NK2; i += stride)  wd2db[i] = (__bf16)wd2d[i];
  for (int i = tid; i < ND * ND; i += stride)   wddb[i]  = (__bf16)wdd[i];
}

// ---------------------------------------------------------------------------
// Wm[b,e] = sum_d m[b,d] * W_ddm[e,d] + b_dm[e].  M=256, N=512, K=512.
// One wave per 16x16 tile; 512 tiles; 8 waves / block -> 64 blocks.
// ---------------------------------------------------------------------------
__global__ void wm_kernel(const __bf16* __restrict__ mb,
                          const __bf16* __restrict__ wb,
                          const float*  __restrict__ bdm,
                          float* __restrict__ wm) {
  int wave = threadIdx.x >> 5;
  int lane = threadIdx.x & 31;
  int tile = blockIdx.x * 8 + wave;       // 0..511
  int mtile = tile >> 5;                  // 0..15
  int ntile = tile & 31;                  // 0..31
  int l15 = lane & 15;
  int hi  = lane >> 4;
  int arow = (mtile << 4) + l15;
  int bcol = (ntile << 4) + l15;
  int ka = hi ? 16 : 0;    // A: lanes 0-15 hold K 0-15, lanes 16-31 hold K 16-31
  int kb = hi ? 16 : 0;    // B: lanes 0-15 hold K 0-15, lanes 16-31 hold K 16-31

  const __bf16* ap = mb + arow * ND;
  const __bf16* bp = wb + bcol * ND;
  f32x8 acc = {};
  for (int ks = 0; ks < ND; ks += 32) {
    bf16x8 a0 = *(const bf16x8*)(ap + ks + ka);
    bf16x8 a1 = *(const bf16x8*)(ap + ks + ka + 8);
    bf16x16 A = SHUF16(a0, a1);
    bf16x8 b0 = *(const bf16x8*)(bp + ks + kb);
    bf16x8 b1 = *(const bf16x8*)(bp + ks + kb + 8);
    bf16x16 Bm = SHUF16(b0, b1);
    acc = __builtin_amdgcn_wmma_f32_16x16x32_bf16(false, A, false, Bm,
                                                  (short)0, acc, false, false);
  }
  float bias = bdm[bcol];
  int rbase = (mtile << 4) + (hi << 3);
  for (int j = 0; j < 8; ++j)
    wm[(rbase + j) * ND + bcol] = acc[j] + bias;
}

// ---------------------------------------------------------------------------
// Fused main chain. grid = 256 (one block per batch b), 256 threads (8 waves).
// Per 32-row tile: async k->LDS, build conc(bf16), GEMM1 (K=1024), scale,
// GEMM2 (K=512), softmax, weighted-sum accumulate.
// ---------------------------------------------------------------------------
__global__ void __launch_bounds__(256)
fused_kernel(const float* __restrict__ c_i,
             const float* __restrict__ k_hw,
             const float* __restrict__ b_dk,
             const float* __restrict__ b_d1,
             const float* __restrict__ b_d2,
             const float* __restrict__ wm,
             const __bf16* __restrict__ wd2db,
             const __bf16* __restrict__ wddb,
             float* __restrict__ out) {
  // phase-disjoint unions (barrier-separated):
  __shared__ __align__(16) char s_u1[MT * ND * 4];   // 64KB: k fp32 -> cI bf16
  __shared__ __align__(16) char s_u2[MT * NK2 * 2];  // 64KB: conc bf16 -> ra fp32
  __shared__ float s_wm[ND], s_bdk[ND], s_b1[ND], s_b2[ND], s_ci[ND];
  __shared__ float s_red[256], s_rowmax[MT], s_rowsum[MT];

  float*  const s_k    = (float*)s_u1;
  __bf16* const s_cI   = (__bf16*)s_u1;
  __bf16* const s_conc = (__bf16*)s_u2;
  float*  const s_ra   = (float*)s_u2;

  const int b    = blockIdx.x;
  const int tid  = threadIdx.x;
  const int wave = tid >> 5;
  const int lane = tid & 31;
  const int l15  = lane & 15;
  const int hi   = lane >> 4;
  const int ka   = hi ? 16 : 0;
  const int kb   = hi ? 16 : 0;

  for (int d = tid; d < ND; d += 256) {
    s_wm[d]  = wm[b * ND + d];
    s_bdk[d] = b_dk[d];
    s_b1[d]  = b_d1[d];
    s_b2[d]  = b_d2[d];
    s_ci[d]  = c_i[b * ND + d];
  }
  float racc0 = 0.f, racc1 = 0.f;
  const float* kbase = k_hw + (size_t)b * NHW * ND;
  __syncthreads();

  for (int t0 = 0; t0 < NHW; t0 += MT) {
    const int nrows = (NHW - t0) < MT ? (NHW - t0) : MT;

    // ---- stage k tile [nrows x 512] fp32 into LDS via async DMA ----
    // nrows*128 b128-transfers; always a multiple of 256 -> uniform EXEC.
    {
      const float* gk = kbase + (size_t)t0 * ND;
      int nvec = nrows * (ND / 4);
      for (int v = tid; v < nvec; v += 256) {
        unsigned long long ga = (unsigned long long)(uintptr_t)(gk + v * 4);
        unsigned la = (unsigned)(uintptr_t)(s_k + v * 4);
        asm volatile("global_load_async_to_lds_b128 %0, %1, off"
                     :: "v"(la), "v"(ga) : "memory");
      }
      asm volatile("s_wait_asynccnt 0x0" ::: "memory");
    }
    __syncthreads();

    // ---- build conc tile [32 x 1024] bf16: [Wm*(k+b_dk) | k] ----
    for (int idx = tid; idx < MT * ND; idx += 256) {
      int r = idx >> 9, d = idx & (ND - 1);
      int row = t0 + r;
      float kv = 0.f, iv = 0.f;
      if (row < NHW) {
        kv = s_k[r * ND + d];
        iv = s_wm[d] * (kv + s_bdk[d]);
      }
      s_conc[r * NK2 + d]      = (__bf16)iv;
      s_conc[r * NK2 + ND + d] = (__bf16)kv;
    }
    __syncthreads();

    // ---- GEMM1: II[32x512] = conc[32x1024] . Wd2d[n,k]; +b_d1, *c_i -> cI ----
    // wave owns 4 ntiles (wave + 8*i) and BOTH mtiles: each B fragment feeds
    // two WMMAs -> halves L2 weight traffic per output row.
    {
      f32x8 acc[4][2] = {};
      for (int ks = 0; ks < NK2; ks += 32) {
        bf16x8 a00 = *(const bf16x8*)(s_conc + l15 * NK2 + ks + ka);
        bf16x8 a01 = *(const bf16x8*)(s_conc + l15 * NK2 + ks + ka + 8);
        bf16x16 A0 = SHUF16(a00, a01);
        bf16x8 a10 = *(const bf16x8*)(s_conc + (16 + l15) * NK2 + ks + ka);
        bf16x8 a11 = *(const bf16x8*)(s_conc + (16 + l15) * NK2 + ks + ka + 8);
        bf16x16 A1 = SHUF16(a10, a11);
        for (int i = 0; i < 4; ++i) {
          int n = ((wave + (i << 3)) << 4) + l15;
          const __bf16* bp = wd2db + n * NK2 + ks + kb;
          __builtin_prefetch((const void*)(bp + 32), 0, 1);
          bf16x16 Bm = SHUF16(*(const bf16x8*)bp, *(const bf16x8*)(bp + 8));
          acc[i][0] = __builtin_amdgcn_wmma_f32_16x16x32_bf16(
              false, A0, false, Bm, (short)0, acc[i][0], false, false);
          acc[i][1] = __builtin_amdgcn_wmma_f32_16x16x32_bf16(
              false, A1, false, Bm, (short)0, acc[i][1], false, false);
        }
      }
      int rbase = hi << 3;
      for (int i = 0; i < 4; ++i) {
        int n = ((wave + (i << 3)) << 4) + l15;
        float cs = s_ci[n], bias = s_b1[n];
        for (int mt = 0; mt < 2; ++mt)
          for (int j = 0; j < 8; ++j)
            s_cI[((mt << 4) + rbase + j) * ND + n] =
                (__bf16)(cs * (acc[i][mt][j] + bias));
      }
    }
    __syncthreads();

    // ---- GEMM2: ra[32x512] = cI[32x512] . Wdd[n,k] + b_d2 (fp32 in LDS) ----
    {
      f32x8 acc[4][2] = {};
      for (int ks = 0; ks < ND; ks += 32) {
        bf16x8 a00 = *(const bf16x8*)(s_cI + l15 * ND + ks + ka);
        bf16x8 a01 = *(const bf16x8*)(s_cI + l15 * ND + ks + ka + 8);
        bf16x16 A0 = SHUF16(a00, a01);
        bf16x8 a10 = *(const bf16x8*)(s_cI + (16 + l15) * ND + ks + ka);
        bf16x8 a11 = *(const bf16x8*)(s_cI + (16 + l15) * ND + ks + ka + 8);
        bf16x16 A1 = SHUF16(a10, a11);
        for (int i = 0; i < 4; ++i) {
          int n = ((wave + (i << 3)) << 4) + l15;
          const __bf16* bp = wddb + n * ND + ks + kb;
          bf16x16 Bm = SHUF16(*(const bf16x8*)bp, *(const bf16x8*)(bp + 8));
          acc[i][0] = __builtin_amdgcn_wmma_f32_16x16x32_bf16(
              false, A0, false, Bm, (short)0, acc[i][0], false, false);
          acc[i][1] = __builtin_amdgcn_wmma_f32_16x16x32_bf16(
              false, A1, false, Bm, (short)0, acc[i][1], false, false);
        }
      }
      int rbase = hi << 3;
      for (int i = 0; i < 4; ++i) {
        int n = ((wave + (i << 3)) << 4) + l15;
        float bias = s_b2[n];
        for (int mt = 0; mt < 2; ++mt)
          for (int j = 0; j < 8; ++j)
            s_ra[((mt << 4) + rbase + j) * ND + n] = acc[i][mt][j] + bias;
      }
    }
    __syncthreads();

    // ---- softmax over D=512 per row: 8 threads/row, 64 contiguous elems ----
    {
      int srow = tid >> 3;      // 0..31
      int g8   = tid & 7;
      float* rowp = s_ra + srow * ND + g8 * 64;
      float mx = -1e30f;
      for (int j = 0; j < 64; ++j) mx = fmaxf(mx, rowp[j]);
      s_red[tid] = mx;
      __syncthreads();
      if (g8 == 0) {
        float m2 = -1e30f;
        for (int q = 0; q < 8; ++q) m2 = fmaxf(m2, s_red[(srow << 3) + q]);
        s_rowmax[srow] = m2;
      }
      __syncthreads();
      float rmx = s_rowmax[srow];
      float sum = 0.f;
      for (int j = 0; j < 64; ++j) {
        float e = __expf(rowp[j] - rmx);
        rowp[j] = e;
        sum += e;
      }
      s_red[tid] = sum;
      __syncthreads();
      if (g8 == 0) {
        float s2 = 0.f;
        for (int q = 0; q < 8; ++q) s2 += s_red[(srow << 3) + q];
        s_rowsum[srow] = s2;
      }
    }
    __syncthreads();

    // ---- r += softmax(ra) * k ; thread owns columns tid, tid+256 ----
    // (k rows are hot in L2/WGP$ from the async stage)
    for (int r = 0; r < nrows; ++r) {
      float inv = 1.f / s_rowsum[r];
      const float* krow = kbase + (size_t)(t0 + r) * ND;
      racc0 += s_ra[r * ND + tid]       * inv * krow[tid];
      racc1 += s_ra[r * ND + tid + 256] * inv * krow[tid + 256];
    }
    __syncthreads();  // protect unions before next tile
  }

  out[b * ND + tid]       = racc0;
  out[b * ND + tid + 256] = racc1;
}

// ---------------------------------------------------------------------------
extern "C" void kernel_launch(void* const* d_in, const int* in_sizes, int n_in,
                              void* d_out, int out_size, void* d_ws, size_t ws_size,
                              hipStream_t stream) {
  const float* c_i   = (const float*)d_in[0];
  const float* m     = (const float*)d_in[1];
  const float* k_hw  = (const float*)d_in[2];
  const float* W_ddm = (const float*)d_in[3];
  const float* b_dm  = (const float*)d_in[4];
  // d_in[5] = W_ddk: unused (reference overwrites its product)
  const float* b_dk  = (const float*)d_in[6];
  const float* W_d2d = (const float*)d_in[7];
  const float* b_d1  = (const float*)d_in[8];
  const float* W_dd  = (const float*)d_in[9];
  const float* b_d2  = (const float*)d_in[10];

  char* ws = (char*)d_ws;
  float*  wm    = (float*)(ws + 0);
  __bf16* mb    = (__bf16*)(ws + 524288);
  __bf16* wddmb = (__bf16*)(ws + 786432);
  __bf16* wd2db = (__bf16*)(ws + 1310720);
  __bf16* wddb  = (__bf16*)(ws + 2359296);

  prep_kernel<<<512, 256, 0, stream>>>(m, W_ddm, W_d2d, W_dd,
                                       mb, wddmb, wd2db, wddb);
  wm_kernel<<<64, 256, 0, stream>>>(mb, wddmb, b_dm, wm);
  fused_kernel<<<NB, 256, 0, stream>>>(c_i, k_hw, b_dk, b_d1, b_d2,
                                       wm, wd2db, wddb, (float*)d_out);
}
